// BasicWindowAttention_57329223467359
// MI455X (gfx1250) — compile-verified, matrix operand order checked by reading, not run on hardware
//
#include <hip/hip_runtime.h>

// ---------------------------------------------------------------------------
// BasicWindowAttention for MI455X (gfx1250, wave32, WMMA bf16)
// B=2048 windows, N=64 tokens, DIM=256, HEADS=8, HEAD_DIM=32
// One block per window, 8 waves, wave w owns head w.
// <=256 VGPR/wave (waves-per-EU=4) and 157.7KB LDS -> 2 blocks per WGP.
// Attention computed transposed (S^T = K.Q^T) so softmax reduces along the
// C-layout M axis: in-lane tree + one shfl_xor(16); P and O stores are
// contiguous packed bf16 pairs.
// ---------------------------------------------------------------------------

#define NTOK   64
#define DIM_   256
#define HEADS_ 8
#define NWIN   2048
#define SCALE_ 0.17677669529663687f   // 32^-0.5

typedef __attribute__((ext_vector_type(16))) __bf16 bf16x16;
typedef __attribute__((ext_vector_type(8)))  __bf16 bf16x8;
typedef __attribute__((ext_vector_type(2)))  __bf16 bf16x2;
typedef __attribute__((ext_vector_type(8)))  float  f32x8;
typedef __attribute__((ext_vector_type(2)))  float  f32x2;

__device__ __forceinline__ __bf16 f2bf(float f) { return (__bf16)f; }

__device__ __forceinline__ void st_bf16x2(__bf16* p, float a, float b) {
    f32x2 v = {a, b};
    *(bf16x2*)p = __builtin_convertvector(v, bf16x2);   // v_cvt_pk_bf16_f32
}

// Load one 16-bit WMMA fragment (16 bf16 per lane): two 16B chunks at p, p+16
// CDNA5 16-bit A/B layout: per lane K-halves {0..7,16..23}/{8..15,24..31}
__device__ __forceinline__ bf16x16 frag_ld(const __bf16* p) {
    bf16x8 lo = *(const bf16x8*)(p);
    bf16x8 hi = *(const bf16x8*)(p + 16);
    return __builtin_shufflevector(lo, hi, 0,1,2,3,4,5,6,7,8,9,10,11,12,13,14,15);
}

__device__ __forceinline__ f32x8 wmma_bf16(bf16x16 a, bf16x16 b, f32x8 c) {
    return __builtin_amdgcn_wmma_f32_16x16x32_bf16(
        false, a, false, b, (short)0, c, false, false);
}

// ---------------------------------------------------------------------------
// Kernel 0: fp32 weights -> bf16; rel_index -> transposed copy (coalesced use)
// ---------------------------------------------------------------------------
__global__ void convert_weights_kernel(const float* __restrict__ qkv_w,
                                       const float* __restrict__ proj_w,
                                       const int* __restrict__ rel,
                                       __bf16* __restrict__ wqkv,
                                       __bf16* __restrict__ wproj,
                                       int* __restrict__ relT) {
    int i = blockIdx.x * 256 + threadIdx.x;          // one float2 per thread
    const int nq2 = 768 * 256 / 2;
    const int np2 = 256 * 256 / 2;
    if (i < nq2) {
        f32x2 v = *(const f32x2*)(qkv_w + 2 * i);
        st_bf16x2(wqkv + 2 * i, v.x, v.y);
    } else if (i < nq2 + np2) {
        int j = i - nq2;
        f32x2 v = *(const f32x2*)(proj_w + 2 * j);
        st_bf16x2(wproj + 2 * j, v.x, v.y);
    } else if (i < nq2 + np2 + NTOK * NTOK) {
        int j = i - nq2 - np2;                       // j = m*64 + n
        relT[j] = rel[(j & 63) * 64 + (j >> 6)];     // relT[m][n] = rel[n][m]
    }
}

// ---------------------------------------------------------------------------
// Kernel 1: fully fused window attention
// Dynamic LDS layout (157728 B; 2 blocks/WGP):
//   [0,1024)          sGeo : 64 x {lon,lat,cos(lat),pad}
//   [1024,33792)      sX   : x bf16 64x256      -> later sHR {hav,rel}[key][query]
//   [33792,66560)     sQ   : bf16 64x256
//   [66560,99328)     sK   : bf16 64x256        -> later sO (bf16 64x256)
//   [99328,132096)    sVt  : bf16 V^T [256][64]
//   [132096,150528)   sPst : per-wave P staging, 8 x (16 x stride-72 bf16)
//   [150528,157728)   sABp : per-head packed bf16 {alpha,beta}, [8][225] u32
// ---------------------------------------------------------------------------
__global__ __launch_bounds__(256, 4) void win_attn_kernel(
    const float* __restrict__ x,      const float* __restrict__ qkv_b,
    const float* __restrict__ proj_b, const float* __restrict__ alpha,
    const float* __restrict__ beta,   const int* __restrict__ relT,
    const __bf16* __restrict__ wqkv,  const __bf16* __restrict__ wproj,
    float* __restrict__ out)
{
    extern __shared__ char smem[];
    float*        sGeo = (float*)smem;
    __bf16*       sX   = (__bf16*)(smem + 1024);
    float*        sHR  = (float*)(smem + 1024);        // {hav, rel bits} pairs
    __bf16*       sQ   = (__bf16*)(smem + 33792);
    __bf16*       sK   = (__bf16*)(smem + 66560);
    __bf16*       sO   = (__bf16*)(smem + 66560);
    __bf16*       sVt  = (__bf16*)(smem + 99328);
    __bf16*       sPstage = (__bf16*)(smem + 132096);
    unsigned*     sABp = (unsigned*)(smem + 150528);

    const int tid   = threadIdx.x;
    const int lane  = tid & 31;
    const int wv    = tid >> 5;                 // wave index == head index
    const int llo   = lane & 15;
    const int mhi   = (lane & 16) ? 8 : 0;      // C-layout: upper half -> M+8
    const int chunk = (lane & 16) ? 8 : 0;      // A/B frag K chunk offset
    const int b     = blockIdx.x;
    const long xbase = (long)b * NTOK * (DIM_ + 2);

    __builtin_prefetch(&x[xbase], 0, 0);        // global_prefetch_b8

    // ---- Phase 1: stage x (bf16 packed) + geo + packed alpha/beta tables ---
    for (int i = tid; i < NTOK * (DIM_ / 2); i += 256) {
        int t = i >> 7, c2 = (i & 127) * 2;
        f32x2 v = *(const f32x2*)(x + xbase + t * 258 + c2);   // 8B aligned
        st_bf16x2(sX + t * 256 + c2, v.x, v.y);
    }
    if (tid < NTOK) {
        float lon = x[xbase + tid * 258 + 256];
        float lat = x[xbase + tid * 258 + 257];
        sGeo[tid * 4 + 0] = lon;
        sGeo[tid * 4 + 1] = lat;
        sGeo[tid * 4 + 2] = __cosf(lat);
    }
    for (int i = tid; i < 225 * HEADS_; i += 256) {
        int h = i / 225, idx = i - h * 225;
        f32x2 v = {alpha[idx * HEADS_ + h], beta[idx * HEADS_ + h]};
        sABp[h * 225 + idx] =
            __builtin_bit_cast(unsigned, __builtin_convertvector(v, bf16x2));
    }
    __syncthreads();

    // ---- Phase 2: QKV GEMM (per-wave head slice), 192 WMMAs/wave -----------
#pragma unroll
    for (int s = 0; s < 3; ++s) {               // 0=Q 1=K 2=V
#pragma unroll
        for (int tn = 0; tn < 2; ++tn) {
            int wrow = s * 256 + wv * 32 + tn * 16 + llo;  // weight row = out col
            bf16x16 bfr[8];
#pragma unroll
            for (int kk = 0; kk < 8; ++kk)
                bfr[kk] = frag_ld(wqkv + wrow * 256 + kk * 32 + chunk);
            float bias = qkv_b[wrow];
            int col = wv * 32 + tn * 16 + llo;
            for (int tm = 0; tm < 4; ++tm) {
                f32x8 acc = {0,0,0,0,0,0,0,0};
#pragma unroll
                for (int kk = 0; kk < 8; ++kk) {
                    bf16x16 af = frag_ld(sX + (tm * 16 + llo) * 256 + kk * 32 + chunk);
                    acc = wmma_bf16(af, bfr[kk], acc);
                }
                if (s == 2) {                   // V stored transposed: packed pairs
#pragma unroll
                    for (int r = 0; r < 8; r += 2) {
                        int row = tm * 16 + r + mhi;
                        st_bf16x2(sVt + col * 64 + row, acc[r] + bias, acc[r + 1] + bias);
                    }
                } else {
                    __bf16* dst = (s == 0) ? sQ : sK;
#pragma unroll
                    for (int r = 0; r < 8; ++r) {
                        int row = tm * 16 + r + mhi;
                        dst[row * 256 + col] = f2bf(acc[r] + bias);
                    }
                }
            }
        }
    }
    __syncthreads();

    // ---- Phase 2.5: haversine+rel table, [key m][query n] layout -----------
    for (int i = tid; i < NTOK * NTOK; i += 256) {
        int m = i >> 6, n = i & 63;             // i = m*64 + n
        float dlat = sGeo[n * 4 + 1] - sGeo[m * 4 + 1];
        float dlon = sGeo[n * 4 + 0] - sGeo[m * 4 + 0];
        float sa = __sinf(dlat * 0.5f);
        float sb = __sinf(dlon * 0.5f);
        float a  = sa * sa + sGeo[n * 4 + 2] * sGeo[m * 4 + 2] * sb * sb;
        a = fminf(fmaxf(a, 0.0f), 1.0f);
        f32x2 hr = {2.0f * asinf(sqrtf(a)), __int_as_float(relT[i])};
        *(f32x2*)(sHR + 2 * i) = hr;            // single ds_store_b64, no conflicts
    }
    __syncthreads();

    // ---- Phase 3: transposed attention, one query tile (tj) at a time ------
    bf16x16 kf[4], vtf[4];
#pragma unroll
    for (int ti = 0; ti < 4; ++ti)              // A frags of K (rows = key tokens)
        kf[ti] = frag_ld(sK + (ti * 16 + llo) * 256 + wv * 32 + chunk);
#pragma unroll
    for (int td = 0; td < 2; ++td)
#pragma unroll
        for (int kk = 0; kk < 2; ++kk)          // A frags of V^T (rows = head dim)
            vtf[td * 2 + kk] = frag_ld(sVt + (wv * 32 + td * 16 + llo) * 64 + kk * 32 + chunk);
    __syncthreads();   // all waves consumed sK -> its region may become sO

    const unsigned* sABh = sABp + wv * 225;
    __bf16* sPst = sPstage + wv * (16 * 72);    // private staging, stride 72

#pragma unroll
    for (int tj = 0; tj < 4; ++tj) {            // query-token tile
        bf16x16 qb = frag_ld(sQ + (tj * 16 + llo) * 256 + wv * 32 + chunk);
        f32x8 Scol[4];
#pragma unroll
        for (int ti = 0; ti < 4; ++ti) {
            f32x8 z = {0,0,0,0,0,0,0,0};
            Scol[ti] = wmma_bf16(kf[ti], qb, z);   // C[key i][query j]
        }
        const int j = tj * 16 + llo;            // this lane's query token
        // bias: element (i = key, j = query); table stored [i][j]
#pragma unroll
        for (int ti = 0; ti < 4; ++ti)
#pragma unroll
            for (int r = 0; r < 8; ++r) {
                int i = ti * 16 + r + mhi;
                f32x2 hr = *(const f32x2*)(sHR + 2 * (i * 64 + j));  // ds_load_b64
                bf16x2 ab = __builtin_bit_cast(bf16x2, sABh[__float_as_int(hr.y)]);
                Scol[ti][r] = Scol[ti][r] * SCALE_ + hr.x * (float)ab[0] + (float)ab[1];
            }
        // softmax over key dim: in-lane tree + one cross-half shuffle
        float mx = Scol[0][0];
#pragma unroll
        for (int ti = 0; ti < 4; ++ti)
#pragma unroll
            for (int r = 0; r < 8; ++r) mx = fmaxf(mx, Scol[ti][r]);
        mx = fmaxf(mx, __shfl_xor(mx, 16));
        float sum = 0.0f;
#pragma unroll
        for (int ti = 0; ti < 4; ++ti)
#pragma unroll
            for (int r = 0; r < 8; ++r) {
                float e = __expf(Scol[ti][r] - mx);
                Scol[ti][r] = e;
                sum += e;
            }
        sum += __shfl_xor(sum, 16);
        float inv = __builtin_amdgcn_rcpf(sum);
        // P slice -> private staging as [jloc][i] (contiguous packed pairs)
#pragma unroll
        for (int ti = 0; ti < 4; ++ti)
#pragma unroll
            for (int r = 0; r < 8; r += 2) {
                int i = ti * 16 + r + mhi;
                st_bf16x2(sPst + llo * 72 + i, Scol[ti][r] * inv, Scol[ti][r + 1] * inv);
            }
        // O^T slice = V^T @ P : A = vtf, B = P (cols = this query tile)
        bf16x16 pb0 = frag_ld(sPst + llo * 72 + 0  + chunk);
        bf16x16 pb1 = frag_ld(sPst + llo * 72 + 32 + chunk);
#pragma unroll
        for (int td = 0; td < 2; ++td) {
            f32x8 acc = {0,0,0,0,0,0,0,0};
            acc = wmma_bf16(vtf[td * 2 + 0], pb0, acc);
            acc = wmma_bf16(vtf[td * 2 + 1], pb1, acc);
            // element (d = M, j = N): store sO[query][head*32+d], packed pairs
#pragma unroll
            for (int r = 0; r < 8; r += 2) {
                int d = td * 16 + r + mhi;
                st_bf16x2(sO + (tj * 16 + llo) * 256 + wv * 32 + d, acc[r], acc[r + 1]);
            }
        }
    }
    __syncthreads();

    // ---- Phase 4: out = O @ Wproj^T + proj_b (wave wv -> cols [wv*32,+32)) -
    float* outb = out + (long)b * NTOK * DIM_;
#pragma unroll
    for (int tn = 0; tn < 2; ++tn) {
        int col = wv * 32 + tn * 16 + llo;
        bf16x16 bfr[8];
#pragma unroll
        for (int kk = 0; kk < 8; ++kk)
            bfr[kk] = frag_ld(wproj + col * 256 + kk * 32 + chunk);
        float pb = proj_b[col];
        for (int tm = 0; tm < 4; ++tm) {
            f32x8 acc = {0,0,0,0,0,0,0,0};
#pragma unroll
            for (int kk = 0; kk < 8; ++kk) {
                bf16x16 af = frag_ld(sO + (tm * 16 + llo) * 256 + kk * 32 + chunk);
                acc = wmma_bf16(af, bfr[kk], acc);
            }
#pragma unroll
            for (int r = 0; r < 8; ++r) {
                int row = tm * 16 + r + mhi;
                outb[row * 256 + col] = acc[r] + pb;   // coalesced fp32 store
            }
        }
    }
}

// ---------------------------------------------------------------------------
extern "C" void kernel_launch(void* const* d_in, const int* in_sizes, int n_in,
                              void* d_out, int out_size, void* d_ws, size_t ws_size,
                              hipStream_t stream) {
    const float* x      = (const float*)d_in[0];
    const float* qkv_w  = (const float*)d_in[1];
    const float* qkv_b  = (const float*)d_in[2];
    const float* proj_w = (const float*)d_in[3];
    const float* proj_b = (const float*)d_in[4];
    const float* alpha  = (const float*)d_in[5];
    const float* beta   = (const float*)d_in[6];
    const int*   rel    = (const int*)d_in[7];

    __bf16* wqkv  = (__bf16*)d_ws;                  // 768*256 bf16
    __bf16* wproj = wqkv + 768 * 256;               // 256*256 bf16
    int*    relT  = (int*)(wproj + 256 * 256);      // 64*64 i32 transposed

    const int nconv = (768 * 256 + 256 * 256) / 2 + NTOK * NTOK;
    convert_weights_kernel<<<(nconv + 255) / 256, 256, 0, stream>>>(
        qkv_w, proj_w, rel, wqkv, wproj, relT);

    const int lds_bytes = 157728;   // 2 blocks fit in the 320KB WGP LDS
    hipFuncSetAttribute(reinterpret_cast<const void*>(&win_attn_kernel),
                        hipFuncAttributeMaxDynamicSharedMemorySize, lds_bytes);

    win_attn_kernel<<<NWIN, 256, lds_bytes, stream>>>(
        x, qkv_b, proj_b, alpha, beta, relT, wqkv, wproj, (float*)d_out);
}
